// TransMIL_78134045048859
// MI455X (gfx1250) — matrix operand order repaired and checked by reading, not compiled, not run on hardware
//
#include <hip/hip_runtime.h>

// ---------------------------------------------------------------------------
// Types for CDNA5 WMMA
// ---------------------------------------------------------------------------
typedef __attribute__((ext_vector_type(16))) _Float16 v16h;
typedef __attribute__((ext_vector_type(8)))  _Float16 v8h;
typedef __attribute__((ext_vector_type(8)))  float    v8f;

#define HEADS 8
#define DH    64
#define MLM   256            // landmark count M
#define LFAC  65             // npd / M
#define NTOK  16384
#define T1    16385          // tokens incl cls
#define NPD   16640          // padded sequence
#define DMODEL 512
#define DQKV  1536

// GEMM tiling
#define BM 128
#define BN 64
#define BK 32
#define LP 40                // padded LDS row stride in halves (bank-conflict free)

// ---------------------------------------------------------------------------
// WMMA GEMM:  C[b] = act( scale * A[b] @ op(B[b]) + bias )
// Operands are f16 in global memory; accumulation f32; output f32 or f16.
//   BT=1  : B is [N,K] row-major (A @ B^T); B staged via async-to-LDS
//   BT=0  : B is [K,N] row-major (A @ B);   B staged synchronously (transpose)
//   OF16  : 1 -> f16 output, 0 -> f32 output
// A is always staged with gfx1250 GLOBAL_LOAD_ASYNC_TO_LDS_B128, double
// buffered in LDS, so global latency overlaps the WMMA stream.
// M % 128 == 0, N % 64 == 0, K % 32 == 0 (true for every call site here).
// ---------------------------------------------------------------------------
template <int BT, int OF16>
__global__ __launch_bounds__(256)
void wmma_gemm(const _Float16* __restrict__ A, long sAb, int lda,
               const _Float16* __restrict__ B, long sBb, int ldb,
               void* __restrict__ Cv, long sCb, int ldc,
               int M, int N, int K,
               const float* __restrict__ bias, float scale, int relu)
{
    __shared__ __align__(16) _Float16 sA[2][BM * LP];
    __shared__ __align__(16) _Float16 sB[2][BN * LP];

    const int tid  = threadIdx.x;
    const int wave = tid >> 5;
    const int lane = tid & 31;
    const int lg   = lane >> 4;      // lane group 0/1
    const int lr   = lane & 15;
    const int wm   = wave & 3;       // 4 waves along M
    const int wn   = wave >> 2;      // 2 waves along N
    const int gm0  = blockIdx.y * BM;
    const int gn0  = blockIdx.x * BN;

    const _Float16* Ab = A + (long)blockIdx.z * sAb;   // uniform -> SGPR base
    const _Float16* Bb = B + (long)blockIdx.z * sBb;

    // Async stage of one K-tile into LDS buffer `buf` (per-wave: 2 A ops, +1 B op if BT)
    auto issue_async = [&](int k0, int buf) {
#pragma unroll
        for (int rnd = 0; rnd < 2; ++rnd) {            // A: 512 chunks of 8 halves
            int c = tid + rnd * 256;
            int r = c >> 2, q = (c & 3) << 3;
            unsigned lo = (unsigned)(size_t)(&sA[buf][r * LP + q]);
            unsigned go = (unsigned)(((long)(gm0 + r) * lda + k0 + q) * 2);
            asm volatile("global_load_async_to_lds_b128 %0, %1, %2"
                         :: "v"(lo), "v"(go), "s"(Ab) : "memory");
        }
        if (BT) {                                      // B: 256 chunks of 8 halves
            int r = tid >> 2, q = (tid & 3) << 3;
            unsigned lo = (unsigned)(size_t)(&sB[buf][r * LP + q]);
            unsigned go = (unsigned)(((long)(gn0 + r) * ldb + k0 + q) * 2);
            asm volatile("global_load_async_to_lds_b128 %0, %1, %2"
                         :: "v"(lo), "v"(go), "s"(Bb) : "memory");
        }
    };

    v8f acc[2][2] = {};
    const int nk = K / BK;
    issue_async(0, 0);

    for (int it = 0; it < nk; ++it) {
        const int k0  = it * BK;
        const int buf = it & 1;
        if (it + 1 < nk) issue_async(k0 + BK, buf ^ 1);   // prefetch next tile
        if (!BT) {                                        // transpose-stage B[k][n] -> sB[n][k]
            for (int t = tid; t < BN * BK; t += 256) {
                int n = t & (BN - 1);
                int k = t >> 6;
                sB[buf][n * LP + k] = Bb[(long)(k0 + k) * ldb + gn0 + n];
            }
        }
        if (it + 1 < nk)
            asm volatile("s_wait_asynccnt %0" :: "n"(BT ? 3 : 2) : "memory");
        else
            asm volatile("s_wait_asynccnt 0" ::: "memory");
        __syncthreads();

        // ---- fragments: A and B both in contiguous k order (k = 16*lg + e) ----
        v16h af[2], bf[2];
#pragma unroll
        for (int mi = 0; mi < 2; ++mi) {
            const _Float16* p = &sA[buf][(wm * 32 + mi * 16 + lr) * LP];
            v8h lo = *(const v8h*)(p + 16 * lg);       // K = 16*lg + e
            v8h hi = *(const v8h*)(p + 16 * lg + 8);   // K = 16*lg + 8 + e
#pragma unroll
            for (int e = 0; e < 8; ++e) { af[mi][e] = lo[e]; af[mi][8 + e] = hi[e]; }
        }
#pragma unroll
        for (int ni = 0; ni < 2; ++ni) {
            const _Float16* p = &sB[buf][(wn * 32 + ni * 16 + lr) * LP + 16 * lg];
            v8h b0 = *(const v8h*)(p);                 // K = 16*lg + e
            v8h b1 = *(const v8h*)(p + 8);
#pragma unroll
            for (int e = 0; e < 8; ++e) { bf[ni][e] = b0[e]; bf[ni][8 + e] = b1[e]; }
        }
#pragma unroll
        for (int mi = 0; mi < 2; ++mi)
#pragma unroll
            for (int ni = 0; ni < 2; ++ni)
                acc[mi][ni] = __builtin_amdgcn_wmma_f32_16x16x32_f16(
                    false, af[mi], false, bf[ni], (short)0, acc[mi][ni], false, false);
        __syncthreads();
    }

    // ---- epilogue: C layout row = 8*lg + vgpr, col = lane&15 ----
    float*    Cf = (float*)Cv    + (long)blockIdx.z * sCb;
    _Float16* Ch = (_Float16*)Cv + (long)blockIdx.z * sCb;
#pragma unroll
    for (int mi = 0; mi < 2; ++mi) {
#pragma unroll
        for (int ni = 0; ni < 2; ++ni) {
            const int c  = gn0 + wn * 32 + ni * 16 + lr;
            const float bv = bias ? bias[c] : 0.0f;
#pragma unroll
            for (int i = 0; i < 8; ++i) {
                const int r = gm0 + wm * 32 + mi * 16 + 8 * lg + i;
                float v = acc[mi][ni][i] * scale + bv;
                if (relu) v = fmaxf(v, 0.0f);
                if (OF16) Ch[(long)r * ldc + c] = (_Float16)v;
                else      Cf[(long)r * ldc + c] = v;
            }
        }
    }
}

// ---------------------------------------------------------------------------
// Helper kernels
// ---------------------------------------------------------------------------
__global__ void k_copy(float* __restrict__ d, const float* __restrict__ s, long n) {
    long i = (long)blockIdx.x * blockDim.x + threadIdx.x;
    if (i < n) d[i] = s[i];
}

__global__ void k_zeroh(_Float16* __restrict__ p, long n) {
    long i = (long)blockIdx.x * blockDim.x + threadIdx.x;
    if (i < n) p[i] = (_Float16)0.0f;
}

// vectorized f32 -> f16 conversion (n multiple of 4)
__global__ void k_cvt(const float* __restrict__ s, _Float16* __restrict__ d, long n4) {
    long i = (long)blockIdx.x * blockDim.x + threadIdx.x;
    if (i >= n4) return;
    float4 v = ((const float4*)s)[i];
    _Float16* o = d + i * 4;
    o[0] = (_Float16)v.x; o[1] = (_Float16)v.y;
    o[2] = (_Float16)v.z; o[3] = (_Float16)v.w;
}

// LayerNorm over D=512 (f32 in, f16 out), one block per row
__global__ void k_layernorm(const float* __restrict__ x, const float* __restrict__ g,
                            const float* __restrict__ b, _Float16* __restrict__ out, int Dd)
{
    __shared__ float red[256];
    const long row = blockIdx.x;
    const float* xr = x + row * Dd;
    float s = 0.f;
    for (int i = threadIdx.x; i < Dd; i += 256) s += xr[i];
    red[threadIdx.x] = s; __syncthreads();
    for (int o = 128; o > 0; o >>= 1) { if (threadIdx.x < o) red[threadIdx.x] += red[threadIdx.x + o]; __syncthreads(); }
    const float mu = red[0] / Dd; __syncthreads();
    float v = 0.f;
    for (int i = threadIdx.x; i < Dd; i += 256) { float d = xr[i] - mu; v += d * d; }
    red[threadIdx.x] = v; __syncthreads();
    for (int o = 128; o > 0; o >>= 1) { if (threadIdx.x < o) red[threadIdx.x] += red[threadIdx.x + o]; __syncthreads(); }
    const float rstd = rsqrtf(red[0] / Dd + 1e-5f);
    _Float16* orow = out + row * Dd;
    for (int i = threadIdx.x; i < Dd; i += 256)
        orow[i] = (_Float16)((xr[i] - mu) * rstd * g[i] + b[i]);
}

// landmark mean-pool: out[h][m][d] = mean_j in[(m*LFAC+j)*DQKV + h*64 + d]
__global__ void k_pool(const _Float16* __restrict__ in, _Float16* __restrict__ out) {
    int idx = blockIdx.x * blockDim.x + threadIdx.x;
    if (idx >= HEADS * MLM * DH) return;
    int d = idx & 63, m = (idx >> 6) & 255, h = idx >> 14;
    const _Float16* src = in + (long)(m * LFAC) * DQKV + h * 64 + d;
    float s = 0.f;
    for (int j = 0; j < LFAC; ++j) s += (float)src[(long)j * DQKV];
    out[idx] = (_Float16)(s * (1.0f / LFAC));
}

// row softmax over C columns (f16 storage, f32 math), one block per row
__global__ void k_softmax(_Float16* __restrict__ X, int C) {
    __shared__ float red[256];
    _Float16* xr = X + (long)blockIdx.x * C;
    float m = -1e30f;
    for (int i = threadIdx.x; i < C; i += 256) m = fmaxf(m, (float)xr[i]);
    red[threadIdx.x] = m; __syncthreads();
    for (int o = 128; o > 0; o >>= 1) { if (threadIdx.x < o) red[threadIdx.x] = fmaxf(red[threadIdx.x], red[threadIdx.x + o]); __syncthreads(); }
    m = red[0]; __syncthreads();
    float s = 0.f;
    for (int i = threadIdx.x; i < C; i += 256) { float e = __expf((float)xr[i] - m); xr[i] = (_Float16)e; s += e; }
    red[threadIdx.x] = s; __syncthreads();
    for (int o = 128; o > 0; o >>= 1) { if (threadIdx.x < o) red[threadIdx.x] += red[threadIdx.x + o]; __syncthreads(); }
    const float inv = 1.0f / red[0];
    for (int i = threadIdx.x; i < C; i += 256) xr[i] = (_Float16)((float)xr[i] * inv);
}

// abs row / col sums of a2 [8][256][256] (f16 in)
__global__ void k_rowsum(const _Float16* __restrict__ X, float* __restrict__ out) {
    int r = blockIdx.x * blockDim.x + threadIdx.x;
    if (r >= HEADS * MLM) return;
    const _Float16* xr = X + (long)r * MLM;
    float s = 0.f;
    for (int j = 0; j < MLM; ++j) s += fabsf((float)xr[j]);
    out[r] = s;
}
__global__ void k_colsum(const _Float16* __restrict__ X, float* __restrict__ out) {
    int idx = blockIdx.x * blockDim.x + threadIdx.x;
    if (idx >= HEADS * MLM) return;
    int h = idx >> 8, c = idx & 255;
    const _Float16* xh = X + (long)h * MLM * MLM;
    float s = 0.f;
    for (int i = 0; i < MLM; ++i) s += fabsf((float)xh[i * MLM + c]);
    out[idx] = s;
}
__global__ void k_maxprep(const float* __restrict__ rs, const float* __restrict__ cs,
                          float* __restrict__ outscale) {
    __shared__ float r1[256], r2[256];
    float m1 = -1e30f, m2 = -1e30f;
    for (int i = threadIdx.x; i < HEADS * MLM; i += 256) { m1 = fmaxf(m1, rs[i]); m2 = fmaxf(m2, cs[i]); }
    r1[threadIdx.x] = m1; r2[threadIdx.x] = m2; __syncthreads();
    for (int o = 128; o > 0; o >>= 1) {
        if (threadIdx.x < o) { r1[threadIdx.x] = fmaxf(r1[threadIdx.x], r1[threadIdx.x + o]);
                               r2[threadIdx.x] = fmaxf(r2[threadIdx.x], r2[threadIdx.x + o]); }
        __syncthreads();
    }
    if (threadIdx.x == 0) outscale[0] = 1.0f / (r1[0] * r2[0]);
}
// z0[h][i][j] = a2[h][j][i] * scale
__global__ void k_tscale(const _Float16* __restrict__ X, _Float16* __restrict__ Z,
                         const float* __restrict__ scale) {
    long idx = (long)blockIdx.x * blockDim.x + threadIdx.x;
    if (idx >= (long)HEADS * MLM * MLM) return;
    int j = idx & 255, i = (int)((idx >> 8) & 255); long h = idx >> 16;
    Z[idx] = (_Float16)((float)X[h * MLM * MLM + (long)j * MLM + i] * scale[0]);
}
// T = c*I - X   (f16)
__global__ void k_diag_sub(const _Float16* __restrict__ X, _Float16* __restrict__ T, float c) {
    long idx = (long)blockIdx.x * blockDim.x + threadIdx.x;
    if (idx >= (long)HEADS * MLM * MLM) return;
    int j = idx & 255, i = (int)((idx >> 8) & 255);
    float v = -(float)X[idx];
    if (i == j) v += c;
    T[idx] = (_Float16)v;
}

// depthwise 33x1 conv over sequence, added into attn[n][h*64+d]  (f16)
__global__ void k_res_conv(const _Float16* __restrict__ v, const float* __restrict__ w,
                           _Float16* __restrict__ out) {
    long idx = (long)blockIdx.x * blockDim.x + threadIdx.x;
    if (idx >= (long)NPD * DMODEL) return;
    int c = (int)(idx & (DMODEL - 1));
    int n = (int)(idx >> 9);
    int h = c >> 6;
    float s = 0.f;
    for (int i = 0; i < 33; ++i) {
        int nn = n + i - 16;
        if (nn >= 0 && nn < NPD) s += (float)v[(long)nn * DQKV + c] * w[h * 33 + i];
    }
    out[idx] = (_Float16)((float)out[idx] + s);
}

// h[tok] += proj[255 + tok]   (drop pad rows, residual add, f32)
__global__ void k_resid(float* __restrict__ h, const float* __restrict__ proj) {
    long idx = (long)blockIdx.x * blockDim.x + threadIdx.x;
    if (idx >= (long)T1 * DMODEL) return;
    h[idx] += proj[(long)255 * DMODEL + idx];
}

// PPEG: fused 7x7 + 5x5 + 3x3 depthwise convs on 128x128 grid + identity (f32)
__global__ void k_ppeg(const float* __restrict__ hin, float* __restrict__ hout,
                       const float* __restrict__ w7, const float* __restrict__ b7,
                       const float* __restrict__ w5, const float* __restrict__ b5,
                       const float* __restrict__ w3, const float* __restrict__ b3) {
    const int pix = blockIdx.x;
    const int y = pix >> 7, x = pix & 127;
    for (int c = threadIdx.x; c < DMODEL; c += 256) {
        float acc = hin[(long)(1 + pix) * DMODEL + c] + b7[c] + b5[c] + b3[c];
        for (int dy = -3; dy <= 3; ++dy)
            for (int dx = -3; dx <= 3; ++dx) {
                int yy = y + dy, xx = x + dx;
                if (yy >= 0 && yy < 128 && xx >= 0 && xx < 128)
                    acc += hin[(long)(1 + yy * 128 + xx) * DMODEL + c] * w7[c * 49 + (dy + 3) * 7 + (dx + 3)];
            }
        for (int dy = -2; dy <= 2; ++dy)
            for (int dx = -2; dx <= 2; ++dx) {
                int yy = y + dy, xx = x + dx;
                if (yy >= 0 && yy < 128 && xx >= 0 && xx < 128)
                    acc += hin[(long)(1 + yy * 128 + xx) * DMODEL + c] * w5[c * 25 + (dy + 2) * 5 + (dx + 2)];
            }
        for (int dy = -1; dy <= 1; ++dy)
            for (int dx = -1; dx <= 1; ++dx) {
                int yy = y + dy, xx = x + dx;
                if (yy >= 0 && yy < 128 && xx >= 0 && xx < 128)
                    acc += hin[(long)(1 + yy * 128 + xx) * DMODEL + c] * w3[c * 9 + (dy + 1) * 3 + (dx + 1)];
            }
        hout[(long)(1 + pix) * DMODEL + c] = acc;
    }
    if (blockIdx.x == 0)
        for (int c = threadIdx.x; c < DMODEL; c += 256) hout[c] = hin[c];  // cls token
}

// final LN(row 0) + 2-way classifier
__global__ void k_head(const float* __restrict__ h, const float* __restrict__ g,
                       const float* __restrict__ b, const float* __restrict__ fw,
                       const float* __restrict__ fb, float* __restrict__ out) {
    __shared__ float red[256];
    __shared__ float o[DMODEL];
    float s = 0.f;
    for (int i = threadIdx.x; i < DMODEL; i += 256) s += h[i];
    red[threadIdx.x] = s; __syncthreads();
    for (int off = 128; off > 0; off >>= 1) { if (threadIdx.x < off) red[threadIdx.x] += red[threadIdx.x + off]; __syncthreads(); }
    const float mu = red[0] / DMODEL; __syncthreads();
    float v = 0.f;
    for (int i = threadIdx.x; i < DMODEL; i += 256) { float d = h[i] - mu; v += d * d; }
    red[threadIdx.x] = v; __syncthreads();
    for (int off = 128; off > 0; off >>= 1) { if (threadIdx.x < off) red[threadIdx.x] += red[threadIdx.x + off]; __syncthreads(); }
    const float rstd = rsqrtf(red[0] / DMODEL + 1e-5f);
    for (int i = threadIdx.x; i < DMODEL; i += 256) o[i] = (h[i] - mu) * rstd * g[i] + b[i];
    __syncthreads();
    for (int cls = 0; cls < 2; ++cls) {
        float p = 0.f;
        for (int i = threadIdx.x; i < DMODEL; i += 256) p += o[i] * fw[cls * DMODEL + i];
        red[threadIdx.x] = p; __syncthreads();
        for (int off = 128; off > 0; off >>= 1) { if (threadIdx.x < off) red[threadIdx.x] += red[threadIdx.x + off]; __syncthreads(); }
        if (threadIdx.x == 0) out[cls] = red[0] + fb[cls];
        __syncthreads();
    }
}

// ---------------------------------------------------------------------------
// Host orchestration
// ---------------------------------------------------------------------------
static inline dim3 gemm_grid(int M, int N, int batch) { return dim3(N / BN, M / BM, batch); }

struct Ws {
    float    *h, *h2, *proj, *rsum, *csum, *sc;
    _Float16 *xh, *wfc1, *wqkv1, *wow1, *wqkv2, *wow2;
    _Float16 *xp, *qkv, *big, *ql, *kl, *a2, *z0, *z1, *xz, *t1, *t2, *a3v, *w2, *attn;
};

static void nystrom_layer(float* cur, const float* ng, const float* nb,
                          const _Float16* wqkv, const _Float16* wow, const float* ob,
                          const float* resw, const Ws& W, hipStream_t s)
{
    const float rs = 0.125f;  // DH^-0.5
    // LN -> padded xp (f16)
    k_zeroh<<<(255 * DMODEL + 255) / 256, 256, 0, s>>>(W.xp, (long)255 * DMODEL);
    k_layernorm<<<T1, 256, 0, s>>>(cur, ng, nb, W.xp + (long)255 * DMODEL, DMODEL);
    // qkv = xp @ qkvw^T  (f16 out)
    wmma_gemm<1, 1><<<gemm_grid(NPD, DQKV, 1), 256, 0, s>>>(
        W.xp, 0L, DMODEL, wqkv, 0L, DMODEL, W.qkv, 0L, DQKV, NPD, DQKV, DMODEL, nullptr, 1.0f, 0);
    // landmarks
    k_pool<<<HEADS * MLM * DH / 256, 256, 0, s>>>(W.qkv, W.ql);
    k_pool<<<HEADS * MLM * DH / 256, 256, 0, s>>>(W.qkv + DMODEL, W.kl);
    // a2 = softmax(rs * q_l @ k_l^T)
    wmma_gemm<1, 1><<<gemm_grid(MLM, MLM, HEADS), 256, 0, s>>>(
        W.ql, (long)MLM * DH, DH, W.kl, (long)MLM * DH, DH, W.a2, (long)MLM * MLM, MLM,
        MLM, MLM, DH, nullptr, rs, 0);
    k_softmax<<<HEADS * MLM, 256, 0, s>>>(W.a2, MLM);
    // pinv init: z = a2^T / (max rowsum * max colsum)
    k_rowsum<<<HEADS, 256, 0, s>>>(W.a2, W.rsum);
    k_colsum<<<HEADS, 256, 0, s>>>(W.a2, W.csum);
    k_maxprep<<<1, 256, 0, s>>>(W.rsum, W.csum, W.sc);
    const long n2 = (long)HEADS * MLM * MLM;
    k_tscale<<<(unsigned)(n2 / 256), 256, 0, s>>>(W.a2, W.z0, W.sc);
    // Newton-Schulz iterations (all f16 operands, f32 accumulate)
    _Float16* zc = W.z0; _Float16* zn = W.z1;
    for (int it = 0; it < 6; ++it) {
        wmma_gemm<0, 1><<<gemm_grid(MLM, MLM, HEADS), 256, 0, s>>>(
            W.a2, n2 / HEADS, MLM, zc, n2 / HEADS, MLM, W.xz, n2 / HEADS, MLM,
            MLM, MLM, MLM, nullptr, 1.0f, 0);
        k_diag_sub<<<(unsigned)(n2 / 256), 256, 0, s>>>(W.xz, W.t1, 7.0f);
        wmma_gemm<0, 1><<<gemm_grid(MLM, MLM, HEADS), 256, 0, s>>>(
            W.xz, n2 / HEADS, MLM, W.t1, n2 / HEADS, MLM, W.t2, n2 / HEADS, MLM,
            MLM, MLM, MLM, nullptr, 1.0f, 0);
        k_diag_sub<<<(unsigned)(n2 / 256), 256, 0, s>>>(W.t2, W.t1, 15.0f);
        wmma_gemm<0, 1><<<gemm_grid(MLM, MLM, HEADS), 256, 0, s>>>(
            W.xz, n2 / HEADS, MLM, W.t1, n2 / HEADS, MLM, W.t2, n2 / HEADS, MLM,
            MLM, MLM, MLM, nullptr, 1.0f, 0);
        k_diag_sub<<<(unsigned)(n2 / 256), 256, 0, s>>>(W.t2, W.t1, 13.0f);
        wmma_gemm<0, 1><<<gemm_grid(MLM, MLM, HEADS), 256, 0, s>>>(
            zc, n2 / HEADS, MLM, W.t1, n2 / HEADS, MLM, zn, n2 / HEADS, MLM,
            MLM, MLM, MLM, nullptr, 0.25f, 0);
        _Float16* tmp = zc; zc = zn; zn = tmp;
    }
    // a3 = softmax(rs * q_l @ k^T)   (into big: [8][256][NPD])
    wmma_gemm<1, 1><<<gemm_grid(MLM, NPD, HEADS), 256, 0, s>>>(
        W.ql, (long)MLM * DH, DH, W.qkv + DMODEL, 64L, DQKV, W.big, (long)MLM * NPD, NPD,
        MLM, NPD, DH, nullptr, rs, 0);
    k_softmax<<<HEADS * MLM, 256, 0, s>>>(W.big, NPD);
    // a3v = a3 @ v
    wmma_gemm<0, 1><<<gemm_grid(MLM, DH, HEADS), 256, 0, s>>>(
        W.big, (long)MLM * NPD, NPD, W.qkv + 2 * DMODEL, 64L, DQKV, W.a3v, (long)MLM * DH, DH,
        MLM, DH, NPD, nullptr, 1.0f, 0);
    // a1 = softmax(rs * q @ k_l^T)   (reuse big: [8][NPD][256])
    wmma_gemm<1, 1><<<gemm_grid(NPD, MLM, HEADS), 256, 0, s>>>(
        W.qkv, 64L, DQKV, W.kl, (long)MLM * DH, DH, W.big, (long)NPD * MLM, MLM,
        NPD, MLM, DH, nullptr, rs, 0);
    k_softmax<<<HEADS * NPD, 256, 0, s>>>(W.big, MLM);
    // w2 = Z @ a3v
    wmma_gemm<0, 1><<<gemm_grid(MLM, DH, HEADS), 256, 0, s>>>(
        zc, n2 / HEADS, MLM, W.a3v, (long)MLM * DH, DH, W.w2, (long)MLM * DH, DH,
        MLM, DH, MLM, nullptr, 1.0f, 0);
    // attn[:, h*64:(h+1)*64] = a1 @ w2
    wmma_gemm<0, 1><<<gemm_grid(NPD, DH, HEADS), 256, 0, s>>>(
        W.big, (long)NPD * MLM, MLM, W.w2, (long)MLM * DH, DH, W.attn, 64L, DMODEL,
        NPD, DH, MLM, nullptr, 1.0f, 0);
    // + depthwise residual conv on v
    k_res_conv<<<(unsigned)((long)NPD * DMODEL / 256), 256, 0, s>>>(W.qkv + 2 * DMODEL, resw, W.attn);
    // proj = attn @ ow^T + ob   (f32 out)
    wmma_gemm<1, 0><<<gemm_grid(NPD, DMODEL, 1), 256, 0, s>>>(
        W.attn, 0L, DMODEL, wow, 0L, DMODEL, W.proj, 0L, DMODEL, NPD, DMODEL, DMODEL, ob, 1.0f, 0);
    // residual (drop 255 pad rows)
    k_resid<<<(unsigned)(((long)T1 * DMODEL + 255) / 256), 256, 0, s>>>(cur, W.proj);
}

static inline void cvt(const float* s, _Float16* d, long n, hipStream_t st) {
    k_cvt<<<(unsigned)((n / 4 + 255) / 256), 256, 0, st>>>(s, d, n / 4);
}

extern "C" void kernel_launch(void* const* d_in, const int* in_sizes, int n_in,
                              void* d_out, int out_size, void* d_ws, size_t ws_size,
                              hipStream_t stream)
{
    (void)in_sizes; (void)n_in; (void)out_size; (void)ws_size;
    const float* x      = (const float*)d_in[0];
    const float* fc1w   = (const float*)d_in[1];
    const float* fc1b   = (const float*)d_in[2];
    const float* clstok = (const float*)d_in[3];
    const float* l1_ng  = (const float*)d_in[4];
    const float* l1_nb  = (const float*)d_in[5];
    const float* l1_qkv = (const float*)d_in[6];
    const float* l1_ow  = (const float*)d_in[7];
    const float* l1_ob  = (const float*)d_in[8];
    const float* l1_res = (const float*)d_in[9];
    const float* p_w7   = (const float*)d_in[10];
    const float* p_b7   = (const float*)d_in[11];
    const float* p_w5   = (const float*)d_in[12];
    const float* p_b5   = (const float*)d_in[13];
    const float* p_w3   = (const float*)d_in[14];
    const float* p_b3   = (const float*)d_in[15];
    const float* l2_ng  = (const float*)d_in[16];
    const float* l2_nb  = (const float*)d_in[17];
    const float* l2_qkv = (const float*)d_in[18];
    const float* l2_ow  = (const float*)d_in[19];
    const float* l2_ob  = (const float*)d_in[20];
    const float* l2_res = (const float*)d_in[21];
    const float* ng     = (const float*)d_in[22];
    const float* nb     = (const float*)d_in[23];
    const float* fc2w   = (const float*)d_in[24];
    const float* fc2b   = (const float*)d_in[25];

    char* base = (char*)d_ws;
    size_t off = 0;
    auto alloc = [&](size_t bytes) -> void* {
        void* p = base + off;
        off += (bytes + 255) & ~(size_t)255;
        return p;
    };
    Ws W;
    W.h     = (float*)alloc((size_t)T1 * DMODEL * 4);
    W.h2    = (float*)alloc((size_t)T1 * DMODEL * 4);
    W.proj  = (float*)alloc((size_t)NPD * DMODEL * 4);
    W.rsum  = (float*)alloc(2048 * 4);
    W.csum  = (float*)alloc(2048 * 4);
    W.sc    = (float*)alloc(64);
    W.xh    = (_Float16*)alloc((size_t)NTOK * 1024 * 2);
    W.wfc1  = (_Float16*)alloc((size_t)DMODEL * 1024 * 2);
    W.wqkv1 = (_Float16*)alloc((size_t)DQKV * DMODEL * 2);
    W.wow1  = (_Float16*)alloc((size_t)DMODEL * DMODEL * 2);
    W.wqkv2 = (_Float16*)alloc((size_t)DQKV * DMODEL * 2);
    W.wow2  = (_Float16*)alloc((size_t)DMODEL * DMODEL * 2);
    W.xp    = (_Float16*)alloc((size_t)NPD * DMODEL * 2);
    W.qkv   = (_Float16*)alloc((size_t)NPD * DQKV * 2);
    W.big   = (_Float16*)alloc((size_t)HEADS * NPD * MLM * 2);
    W.ql    = (_Float16*)alloc((size_t)HEADS * MLM * DH * 2);
    W.kl    = (_Float16*)alloc((size_t)HEADS * MLM * DH * 2);
    W.a2    = (_Float16*)alloc((size_t)HEADS * MLM * MLM * 2);
    W.z0    = (_Float16*)alloc((size_t)HEADS * MLM * MLM * 2);
    W.z1    = (_Float16*)alloc((size_t)HEADS * MLM * MLM * 2);
    W.xz    = (_Float16*)alloc((size_t)HEADS * MLM * MLM * 2);
    W.t1    = (_Float16*)alloc((size_t)HEADS * MLM * MLM * 2);
    W.t2    = (_Float16*)alloc((size_t)HEADS * MLM * MLM * 2);
    W.a3v   = (_Float16*)alloc((size_t)HEADS * MLM * DH * 2);
    W.w2    = (_Float16*)alloc((size_t)HEADS * MLM * DH * 2);
    W.attn  = (_Float16*)alloc((size_t)NPD * DMODEL * 2);

    // one-time f32 -> f16 conversions of WMMA operands
    cvt(x,      W.xh,    (long)NTOK * 1024, stream);
    cvt(fc1w,   W.wfc1,  (long)DMODEL * 1024, stream);
    cvt(l1_qkv, W.wqkv1, (long)DQKV * DMODEL, stream);
    cvt(l1_ow,  W.wow1,  (long)DMODEL * DMODEL, stream);
    cvt(l2_qkv, W.wqkv2, (long)DQKV * DMODEL, stream);
    cvt(l2_ow,  W.wow2,  (long)DMODEL * DMODEL, stream);

    // h = [cls ; relu(x @ fc1^T + b)]   (f32 residual stream)
    k_copy<<<2, 256, 0, stream>>>(W.h, clstok, DMODEL);
    wmma_gemm<1, 0><<<gemm_grid(NTOK, DMODEL, 1), 256, 0, stream>>>(
        W.xh, 0L, 1024, W.wfc1, 0L, 1024, W.h + DMODEL, 0L, DMODEL,
        NTOK, DMODEL, 1024, fc1b, 1.0f, 1);

    // layer 1
    nystrom_layer(W.h, l1_ng, l1_nb, W.wqkv1, W.wow1, l1_ob, l1_res, W, stream);
    // PPEG: h -> h2
    k_ppeg<<<NTOK, 256, 0, stream>>>(W.h, W.h2, p_w7, p_b7, p_w5, p_b5, p_w3, p_b3);
    // layer 2 (on h2)
    nystrom_layer(W.h2, l2_ng, l2_nb, W.wqkv2, W.wow2, l2_ob, l2_res, W, stream);
    // head
    k_head<<<1, 256, 0, stream>>>(W.h2, ng, nb, fc2w, fc2b, (float*)d_out);
}
